// EW_MHSA_Hybrid_3951369912672
// MI455X (gfx1250) — matrix operand order repaired and checked by reading, not compiled, not run on hardware
//
#include <hip/hip_runtime.h>
#include <hip/hip_bf16.h>
#include <cstddef>
#include <cstdint>

// ---- problem constants (from reference) ----
#define B_    32
#define CIN   256
#define HH    56
#define WW    56
#define HW    3136          // 56*56
#define NOUT  768           // 512 qk channels + 256 v channels (concatenated GEMM)
#define WS_   7
#define NWIN  64            // 8x8 windows
#define HEADS 4
#define DH    64
#define NTOK  49            // 7*7 tokens per window (padded to 64)

typedef __bf16 bf16_t;
typedef __attribute__((ext_vector_type(16))) __bf16 v16bf;
typedef __attribute__((ext_vector_type(8)))  float  v8f;

__device__ __forceinline__ bf16_t f2bf(float x) {
  unsigned int u = __float_as_uint(x);
  // round-to-nearest-even truncation to bf16
  unsigned int r = (u + 0x7fffu + ((u >> 16) & 1u)) >> 16;
  unsigned short s = (unsigned short)r;
  bf16_t b;
  __builtin_memcpy(&b, &s, 2);
  return b;
}

// LDS byte offset of a pointer to __shared__ (low 32 bits of the generic address)
__device__ __forceinline__ unsigned lds_off(const void* p) {
  return (unsigned)(uintptr_t)p;
}

// ============================================================================
// Kernel 1: fused 1x1 convs as GEMM with WMMA bf16.
//   ws[b, hw, n] = sum_k x[b, k, hw] * w[n, k]   (n<512: qk_w ; n>=512: relu(v_w))
// Each wave computes a 16(M) x 64(N) tile: one A fragment feeds 4 WMMAs.
// Block 256 = 8 waves as 2(M) x 4(N) -> 32 x 256 block tile.
// grid (3136/32, 768/256, 32).
// ============================================================================
__global__ void conv_qkv_wmma(const float* __restrict__ x,
                              const float* __restrict__ qk_w,
                              const float* __restrict__ v_w,
                              bf16_t* __restrict__ ws) {
  const int lane = threadIdx.x & 31;
  const int wave = threadIdx.x >> 5;            // 0..7
  const int ln   = lane & 15;
  const int hf   = lane >> 4;                   // half of wave (0/1)

  const int mTile = blockIdx.x * 32 + (wave & 1) * 16;     // spatial tile base
  const int nBase = blockIdx.y * 256 + (wave >> 1) * 64;   // out-ch group base
  const int b     = blockIdx.z;

  const int m = mTile + ln;                     // A-matrix row (spatial index)

  // per-N-tile weight row pointers (row = out channel, contiguous K)
  const float* wrow[4];
#pragma unroll
  for (int nt = 0; nt < 4; ++nt) {
    const int n = nBase + nt * 16 + ln;
    wrow[nt] = (n < 512) ? (qk_w + (size_t)n * CIN)
                         : (v_w + (size_t)(n - 512) * CIN);
  }
  const float* xcol = x + (size_t)b * CIN * HW + m;        // + k*HW walks channels

  v8f acc[4] = {};
  for (int kb = 0; kb < CIN; kb += 32) {
    // prefetch next K-step of the A column (global_prefetch_b8)
    if (kb + 32 < CIN)
      __builtin_prefetch(xcol + (size_t)(kb + 32 + (hf ? 8 : 0)) * HW, 0, 1);

    v16bf af;
#pragma unroll
    for (int j = 0; j < 16; ++j) {
      // A operand, k order chosen here (same as B): k = j + 16*half
      const int ka = kb + j + (hf ? 16 : 0);
      af[j] = f2bf(xcol[(size_t)ka * HW]);
    }
#pragma unroll
    for (int nt = 0; nt < 4; ++nt) {
      v16bf bf;
#pragma unroll
      for (int j = 0; j < 16; ++j) {
        // B operand, k order chosen here (same as A): k = j + 16*half
        bf[j] = f2bf(wrow[nt][kb + j + (hf ? 16 : 0)]);
      }
      acc[nt] = __builtin_amdgcn_wmma_f32_16x16x32_bf16(false, af, false, bf,
                                                        (short)0, acc[nt], false, false);
    }
  }

  const bool relu = (nBase >= 512);             // tile-uniform (boundary aligned)
#pragma unroll
  for (int nt = 0; nt < 4; ++nt) {
    const int n = nBase + nt * 16 + ln;
#pragma unroll
    for (int v = 0; v < 8; ++v) {
      const int mr = mTile + v + 8 * hf;        // C/D layout: m = v + 8*half
      float val = acc[nt][v];
      if (relu && val < 0.f) val = 0.f;
      ws[((size_t)b * HW + mr) * NOUT + n] = f2bf(val);
    }
  }
}

// ============================================================================
// Kernel 2: windowed MHSA for one (batch, window, head) per block.
//   part==0: close (contiguous 7x7); part==1: remote (dilation-8 grid)
//   accumulate==0: store; ==1: out += (stream order sums the two partitions)
// block = 128 threads (4 waves); wave w owns query rows [16w, 16w+16).
// LDS: Q/K/V/P tiles, 64x64 bf16 each = 32 KB.
// Q/K/V rows are fetched with async global->LDS b128 transfers (ASYNCcnt).
// ============================================================================
__global__ void win_attn_wmma(const bf16_t* __restrict__ ws,
                              float* __restrict__ out,
                              int part, int accumulate) {
  __shared__ bf16_t q_lds[64][64];
  __shared__ bf16_t k_lds[64][64];
  __shared__ bf16_t v_lds[64][64];
  __shared__ bf16_t p_lds[64][64];

  const int b    = blockIdx.z;
  const int head = blockIdx.y;
  const int wy   = blockIdx.x >> 3;
  const int wx   = blockIdx.x & 7;
  const int tid  = threadIdx.x;

  auto pix = [&](int t) -> int {
    const int th = t / WS_, tw = t % WS_;
    int h, w;
    if (part == 0) { h = wy * WS_ + th; w = wx * WS_ + tw; }   // close
    else           { h = th * 8 + wy;   w = tw * 8 + wx;   }   // remote (dilated)
    return h * WW + w;
  };

  // ---- async global->LDS: 16B chunks of contiguous 64-ch rows ----
  // q at +0B, k at +512B (256 bf16), v at +1024B (512 bf16) from the q base.
  for (int idx = tid; idx < NTOK * 8; idx += 128) {
    const int t = idx >> 3, c8 = idx & 7;
    const unsigned goff = (unsigned)(((((size_t)b * HW + pix(t)) * NOUT) +
                                      head * DH + c8 * 8) * sizeof(bf16_t));
    const unsigned lq = lds_off(&q_lds[t][c8 * 8]);
    const unsigned lk = lds_off(&k_lds[t][c8 * 8]);
    const unsigned lv = lds_off(&v_lds[t][c8 * 8]);
    asm volatile(
        "global_load_async_to_lds_b128 %0, %3, %4\n\t"
        "global_load_async_to_lds_b128 %1, %3, %4 offset:512\n\t"
        "global_load_async_to_lds_b128 %2, %3, %4 offset:1024"
        :: "v"(lq), "v"(lk), "v"(lv), "v"(goff), "s"(ws)
        : "memory");
  }
  // zero-pad token rows 49..63 (disjoint from async-written rows)
  for (int idx = tid; idx < (64 - NTOK) * 8; idx += 128) {
    const int t = NTOK + (idx >> 3), c8 = idx & 7;
    const uint4 z4 = {0u, 0u, 0u, 0u};
    *(uint4*)&q_lds[t][c8 * 8] = z4;
    *(uint4*)&k_lds[t][c8 * 8] = z4;
    *(uint4*)&v_lds[t][c8 * 8] = z4;
  }
  asm volatile("s_wait_asynccnt 0x0" ::: "memory");
  __syncthreads();

  const int lane = tid & 31, wv = tid >> 5;
  const int ln = lane & 15, hf = lane >> 4;

  // ---- S = Q K^T  (M=16 queries of this wave, N=64 keys, K=64 dims) ----
  v16bf qa[2];
#pragma unroll
  for (int ks = 0; ks < 2; ++ks)
#pragma unroll
    for (int j = 0; j < 16; ++j) {
      const int d = ks * 32 + (j < 8 ? j : j + 8) + (hf ? 8 : 0);
      qa[ks][j] = q_lds[wv * 16 + ln][d];
    }

  v8f sacc[4] = {};
#pragma unroll
  for (int nt = 0; nt < 4; ++nt)
#pragma unroll
    for (int ks = 0; ks < 2; ++ks) {
      v16bf kf;
#pragma unroll
      for (int j = 0; j < 16; ++j) {
        const int d = ks * 32 + (j < 8 ? j : j + 8) + (hf ? 8 : 0);
        kf[j] = k_lds[nt * 16 + ln][d];         // B[k=d][n=key] = K[key][d]
      }
      sacc[nt] = __builtin_amdgcn_wmma_f32_16x16x32_bf16(false, qa[ks], false, kf,
                                                         (short)0, sacc[nt], false, false);
    }

  // ---- softmax over keys (scale 1/8), P -> LDS as bf16 ----
#pragma unroll
  for (int v = 0; v < 8; ++v) {
    float sv[4], mx = -1e30f;
#pragma unroll
    for (int nt = 0; nt < 4; ++nt) {
      const int key = nt * 16 + ln;
      sv[nt] = (key < NTOK) ? sacc[nt][v] * 0.125f : -1e30f;
      mx = fmaxf(mx, sv[nt]);
    }
#pragma unroll
    for (int mk = 1; mk < 16; mk <<= 1) mx = fmaxf(mx, __shfl_xor(mx, mk, 32));
    float e[4], sum = 0.f;
#pragma unroll
    for (int nt = 0; nt < 4; ++nt) {
      const int key = nt * 16 + ln;
      e[nt] = (key < NTOK) ? __expf(sv[nt] - mx) : 0.f;
      sum += e[nt];
    }
#pragma unroll
    for (int mk = 1; mk < 16; mk <<= 1) sum += __shfl_xor(sum, mk, 32);
    const float r = 1.f / sum;
    const int row = wv * 16 + v + 8 * hf;       // C/D layout row
#pragma unroll
    for (int nt = 0; nt < 4; ++nt)
      p_lds[row][nt * 16 + ln] = f2bf(e[nt] * r);
  }
  // wave reads back only its own P rows -> no cross-wave barrier needed

  // ---- O = P V  (M=16 queries, N=64 dims, K=64 keys) ----
  v16bf pa[2];
#pragma unroll
  for (int ks = 0; ks < 2; ++ks)
#pragma unroll
    for (int j = 0; j < 16; ++j) {
      const int kk = ks * 32 + (j < 8 ? j : j + 8) + (hf ? 8 : 0);
      pa[ks][j] = p_lds[wv * 16 + ln][kk];
    }

  v8f oacc[4] = {};
#pragma unroll
  for (int dt = 0; dt < 4; ++dt)
#pragma unroll
    for (int ks = 0; ks < 2; ++ks) {
      v16bf vf;
#pragma unroll
      for (int j = 0; j < 16; ++j) {
        const int key = ks * 32 + (j < 8 ? j : j + 8) + (hf ? 8 : 0);
        vf[j] = v_lds[key][dt * 16 + ln];       // B[k=key][n=d] = V[key][d]
      }
      oacc[dt] = __builtin_amdgcn_wmma_f32_16x16x32_bf16(false, pa[ks], false, vf,
                                                         (short)0, oacc[dt], false, false);
    }

  // ---- write output (channel-major, per reference layout) ----
#pragma unroll
  for (int dt = 0; dt < 4; ++dt)
#pragma unroll
    for (int v = 0; v < 8; ++v) {
      const int q = wv * 16 + v + 8 * hf;
      if (q < NTOK) {
        const int c = head * DH + dt * 16 + ln;
        const size_t o = ((size_t)b * CIN + c) * HW + pix(q);
        const float val = oacc[dt][v];
        if (accumulate) out[o] += val;
        else            out[o]  = val;
      }
    }
}

// ============================================================================
// launcher
// ============================================================================
extern "C" void kernel_launch(void* const* d_in, const int* in_sizes, int n_in,
                              void* d_out, int out_size, void* d_ws, size_t ws_size,
                              hipStream_t stream) {
  (void)in_sizes; (void)n_in; (void)out_size; (void)ws_size;
  const float* x    = (const float*)d_in[0];   // (32,256,56,56)
  const float* qk_w = (const float*)d_in[1];   // (512,256)
  const float* v_w  = (const float*)d_in[2];   // (256,256)
  float* out = (float*)d_out;                  // (32,256,56,56)
  bf16_t* ws = (bf16_t*)d_ws;                  // [32][3136][768] bf16 (~147 MB)

  // fused qk + relu(v) 1x1 convs as a single WMMA GEMM
  conv_qkv_wmma<<<dim3(HW / 32, NOUT / 256, B_), 256, 0, stream>>>(x, qk_w, v_w, ws);

  // close partition: plain store; remote partition: accumulate (stream-ordered)
  win_attn_wmma<<<dim3(NWIN, HEADS, B_), 128, 0, stream>>>(ws, out, 0, 0);
  win_attn_wmma<<<dim3(NWIN, HEADS, B_), 128, 0, stream>>>(ws, out, 1, 1);
}
